// caSchNetEncoder_34505767256268
// MI455X (gfx1250) — matrix operand order repaired and checked by reading, not compiled, not run on hardware
//
#include <hip/hip_runtime.h>
#include <math.h>

#define N_NODES   20000
#define N_EDGES   640000
#define HID       128
#define IN_DIM    5
#define L_LAYERS  7
#define FILT_LAYERS 4
#define CUTOFF_F  10.0f
// 128 + 8 bf16 pad -> 272B row stride: lane r of a fragment load hits dword
// banks 4r..4r+3 -> conflict-free b128 LDS reads across the 16 lanes.
#define LDS_STRIDE 136

typedef __attribute__((ext_vector_type(16))) __bf16   v16bf;
typedef __attribute__((ext_vector_type(8)))  float    v8f;
typedef __attribute__((ext_vector_type(4)))  unsigned int u32x4;

union Frag { v16bf v; u32x4 u[2]; };

__device__ __forceinline__ unsigned short f2bf(float f) {
  unsigned int u = __float_as_uint(f);
  u += 0x7fffu + ((u >> 16) & 1u);          // round-to-nearest-even
  return (unsigned short)(u >> 16);
}

// tanh-gelu, branch-free on HW transcendentals:
//   0.5*x*(1+tanh(k*(x+0.044715x^3))) == x * sigmoid(2k*(x+0.044715x^3))
//   == x / (1 + exp2(A*x + B*x^3)),  A = -2*log2(e)*k, B = A*0.044715
// exp2 overflow/underflow gives the correct +/- saturation limits for free.
__device__ __forceinline__ float gelu_tanh(float x) {
  const float A = -2.8853900817779268f * 0.7978845608028654f;
  const float B = A * 0.044715f;
  float x2 = x * x;
  float p  = x * __builtin_fmaf(B, x2, A);
  float e  = __builtin_amdgcn_exp2f(p);       // v_exp_f32 (TRANS)
  float r  = __builtin_amdgcn_rcpf(1.0f + e); // v_rcp_f32 (TRANS)
  return x * r;
}

// Batched B-fragment load (8x ds_load_b128) + 4-step K chain.
__device__ __forceinline__ v8f gemm_row_tile(const Frag a[4],
                                             const unsigned short* __restrict__ bufW,
                                             int nt, int mrow, int hi) {
  Frag b[4];
#pragma unroll
  for (int kt = 0; kt < 4; ++kt) {
    const unsigned short* q = bufW + (nt * 16 + mrow) * LDS_STRIDE + kt * 32 + hi * 8;
    b[kt].u[0] = *(const u32x4*)(q);
    b[kt].u[1] = *(const u32x4*)(q + 16);
  }
  v8f acc = {};
#pragma unroll
  for (int kt = 0; kt < 4; ++kt)
    acc = __builtin_amdgcn_wmma_f32_16x16x32_bf16(
        false, a[kt].v, false, b[kt].v, (short)0, acc, false, false);
  return acc;
}

// ---------------------------------------------------------------------------
// Weight prep: f32 [nmats][128][128] (k-major) -> bf16 [nmats][128][128]
// transposed (n-major), so a B-fragment lane (column n) reads two runs of 8 contiguous
// K values, matching the ISA 16-bit B-matrix layout.
// ---------------------------------------------------------------------------
__global__ __launch_bounds__(256) void transpose_to_bf16(
    const float* __restrict__ src, unsigned short* __restrict__ dst, int nmats) {
  int i = blockIdx.x * 256 + threadIdx.x;
  if (i >= nmats * HID * HID) return;
  int mat = i >> 14;
  int r   = i & 16383;
  int n   = r >> 7;
  int k   = r & 127;
  dst[i] = f2bf(src[(size_t)mat * HID * HID + (size_t)k * HID + n]);
}

// h = z[:, :IN_DIM] @ emblin_W + emblin_b + z[:, IN_DIM:]
__global__ __launch_bounds__(256) void embed_kernel(
    const float* __restrict__ z, const float* __restrict__ eW,
    const float* __restrict__ eb, float* __restrict__ h) {
  int i = blockIdx.x * 256 + threadIdx.x;
  if (i >= N_NODES * HID) return;
  int n = i >> 7, c = i & 127;
  const float* zr = z + (size_t)n * (IN_DIM + HID);
  float acc = eb[c] + zr[IN_DIM + c];
#pragma unroll
  for (int k = 0; k < IN_DIM; ++k) acc += zr[k] * eW[k * HID + c];
  h[i] = acc;
}

__global__ __launch_bounds__(256) void zero_kernel(float* __restrict__ p, int n) {
  int i = blockIdx.x * 256 + threadIdx.x;
  if (i < n) p[i] = 0.0f;
}

// ---------------------------------------------------------------------------
// Fused edge pipeline: 128 edges/block.
//   W = edge_attr; 4x (W = W@filt_Wk + bk, gelu between); W *= cutoff;
//   msg = xf[src]*W; atomicAdd agg[dst].
// All 4 GEMMs stay in one bf16 LDS tile (in-place: each wave's A fragments
// capture its full 16x128 strip before it writes back).
// ---------------------------------------------------------------------------
__global__ __launch_bounds__(256) void edge_filter_scatter(
    const float* __restrict__ edge_attr,        // [E][128]
    const int*   __restrict__ edge_index,       // [2][E] (src row 0, dst row 1)
    const float* __restrict__ edge_len,         // [E]
    const unsigned short* __restrict__ wfilt,   // [4][128][128] bf16, n-major
    const float* __restrict__ bfilt,            // [4][128]
    const float* __restrict__ xf,               // [N][128]
    float*       __restrict__ agg) {            // [N][128]
  extern __shared__ unsigned short smem[];
  unsigned short* bufA = smem;                     // 128 x LDS_STRIDE (activations)
  unsigned short* bufW = smem + 128 * LDS_STRIDE;  // 128 x LDS_STRIDE (weights)

  const int tid  = threadIdx.x;
  const int lane = tid & 31;
  const int wave = tid >> 5;
  const int mrow = lane & 15;
  const int hi   = lane >> 4;
  const int r0   = wave * 16;
  const int e0   = blockIdx.x * 128;               // N_EDGES % 128 == 0

  // edge_attr tile -> bf16 LDS (coalesced f32 reads)
  for (int i = tid; i < 128 * 128; i += 256) {
    int r = i >> 7, c = i & 127;
    bufA[r * LDS_STRIDE + c] = f2bf(edge_attr[(size_t)(e0 + r) * HID + c]);
  }

  for (int kl = 0; kl < FILT_LAYERS; ++kl) {
    __syncthreads();  // prev bufW reads + bufA writes done
    const unsigned short* wsrc = wfilt + kl * (HID * HID);
    for (int i = tid; i < HID * HID; i += 256)
      bufW[(i >> 7) * LDS_STRIDE + (i & 127)] = wsrc[i];
    __syncthreads();

    // A fragments: full 16x128 strip for this wave (enables in-place update)
    Frag a[4];
    {
      const unsigned short* ar = bufA + (r0 + mrow) * LDS_STRIDE;
#pragma unroll
      for (int kt = 0; kt < 4; ++kt) {
        const unsigned short* p = ar + kt * 32 + hi * 8;  // K groups of 8 per half-wave
        a[kt].u[0] = *(const u32x4*)(p);
        a[kt].u[1] = *(const u32x4*)(p + 16);
      }
    }

    // hoist per-wave bias values out of the nt loop
    float bb[8];
    {
      const float* bias = bfilt + kl * HID;
#pragma unroll
      for (int nt = 0; nt < 8; ++nt) bb[nt] = bias[nt * 16 + mrow];
    }

    if (kl < FILT_LAYERS - 1) {
#pragma unroll
      for (int nt = 0; nt < 8; ++nt) {
        v8f acc = gemm_row_tile(a, bufW, nt, mrow, hi);
        const int col = nt * 16 + mrow;
#pragma unroll
        for (int v = 0; v < 8; ++v)
          bufA[(r0 + hi * 8 + v) * LDS_STRIDE + col] = f2bf(gelu_tanh(acc[v] + bb[nt]));
      }
    } else {
      // final filter layer: cutoff gate + gather xf[src] + scatter-add agg[dst]
      float cm[8]; int es[8]; int ed[8];
#pragma unroll
      for (int v = 0; v < 8; ++v) {
        int e = e0 + r0 + hi * 8 + v;
        cm[v] = (edge_len[e] <= CUTOFF_F) ? 1.0f : 0.0f;
        es[v] = edge_index[e];
        ed[v] = edge_index[N_EDGES + e];
      }
#pragma unroll
      for (int nt = 0; nt < 8; ++nt) {
        v8f acc = gemm_row_tile(a, bufW, nt, mrow, hi);
        const int col = nt * 16 + mrow;
#pragma unroll
        for (int v = 0; v < 8; ++v) {
          if (cm[v] != 0.0f) {
            float w = acc[v] + bb[nt];
            float m = w * xf[(size_t)es[v] * HID + col];
            atomicAdd(&agg[(size_t)ed[v] * HID + col], m);
          }
        }
      }
    }
  }
}

// ---------------------------------------------------------------------------
// Node-side GEMM: Y = op(X @ W + b) [optionally Y += ...], 128 rows/block.
// ---------------------------------------------------------------------------
#define NG_BIAS  1
#define NG_GELU  2
#define NG_ACCUM 4

__global__ __launch_bounds__(256) void node_gemm128(
    const float* __restrict__ X,                // [nrows][128]
    const unsigned short* __restrict__ Wt,      // [128][128] bf16, n-major
    const float* __restrict__ bias,             // [128] (if NG_BIAS)
    float* __restrict__ Y,                      // [nrows][128]
    int nrows, int flags) {
  extern __shared__ unsigned short smem[];
  unsigned short* bufA = smem;
  unsigned short* bufW = smem + 128 * LDS_STRIDE;

  const int tid  = threadIdx.x;
  const int lane = tid & 31;
  const int wave = tid >> 5;
  const int mrow = lane & 15;
  const int hi   = lane >> 4;
  const int r0   = wave * 16;
  const int n0   = blockIdx.x * 128;

  for (int i = tid; i < 128 * 128; i += 256) {
    int r = i >> 7, c = i & 127;
    float xv = (n0 + r < nrows) ? X[(size_t)(n0 + r) * HID + c] : 0.0f;
    bufA[r * LDS_STRIDE + c] = f2bf(xv);
  }
  for (int i = tid; i < HID * HID; i += 256)
    bufW[(i >> 7) * LDS_STRIDE + (i & 127)] = Wt[i];
  __syncthreads();

  Frag a[4];
  {
    const unsigned short* ar = bufA + (r0 + mrow) * LDS_STRIDE;
#pragma unroll
    for (int kt = 0; kt < 4; ++kt) {
      const unsigned short* p = ar + kt * 32 + hi * 8;
      a[kt].u[0] = *(const u32x4*)(p);
      a[kt].u[1] = *(const u32x4*)(p + 16);
    }
  }

  float bb[8];
#pragma unroll
  for (int nt = 0; nt < 8; ++nt)
    bb[nt] = (flags & NG_BIAS) ? bias[nt * 16 + mrow] : 0.0f;

#pragma unroll
  for (int nt = 0; nt < 8; ++nt) {
    v8f acc = gemm_row_tile(a, bufW, nt, mrow, hi);
    const int col = nt * 16 + mrow;
#pragma unroll
    for (int v = 0; v < 8; ++v) {
      int row = n0 + r0 + hi * 8 + v;
      if (row < nrows) {
        float val = acc[v] + bb[nt];
        if (flags & NG_GELU) val = gelu_tanh(val);
        size_t o = (size_t)row * HID + col;
        if (flags & NG_ACCUM) Y[o] += val; else Y[o] = val;
      }
    }
  }
}

// ---------------------------------------------------------------------------
extern "C" void kernel_launch(void* const* d_in, const int* in_sizes, int n_in,
                              void* d_out, int out_size, void* d_ws, size_t ws_size,
                              hipStream_t stream) {
  const float* z        = (const float*)d_in[0];
  const int*   eidx     = (const int*)d_in[1];
  const float* elen     = (const float*)d_in[2];
  const float* eattr    = (const float*)d_in[3];
  const float* emblin_W = (const float*)d_in[4];
  const float* emblin_b = (const float*)d_in[5];
  const float* filt_W   = (const float*)d_in[6];
  const float* filt_b   = (const float*)d_in[7];
  const float* lin1_W   = (const float*)d_in[8];
  const float* lin2_W   = (const float*)d_in[9];
  const float* lin2_b   = (const float*)d_in[10];
  const float* lin_W    = (const float*)d_in[11];
  const float* lin_b    = (const float*)d_in[12];

  float* h = (float*)d_out;                       // [N][128], final output

  char* ws = (char*)d_ws;
  float* xf  = (float*)ws;                                        // N*128 f32 (xf, then x)
  float* agg = (float*)(ws + (size_t)N_NODES * HID * 4);          // N*128 f32
  unsigned short* wFilt = (unsigned short*)(ws + (size_t)2 * N_NODES * HID * 4);
  unsigned short* wLin1 = wFilt + (size_t)L_LAYERS * FILT_LAYERS * HID * HID;
  unsigned short* wLin2 = wLin1 + (size_t)L_LAYERS * HID * HID;
  unsigned short* wLin  = wLin2 + (size_t)L_LAYERS * HID * HID;

  const int GEMM_LDS = 2 * 128 * LDS_STRIDE * (int)sizeof(unsigned short); // 69632 B

  // one-time-per-launch weight transpose + bf16 convert (tiny: ~3.2 MB read)
  {
    int nm = L_LAYERS * FILT_LAYERS;
    transpose_to_bf16<<<(nm * HID * HID + 255) / 256, 256, 0, stream>>>(filt_W, wFilt, nm);
    nm = L_LAYERS;
    transpose_to_bf16<<<(nm * HID * HID + 255) / 256, 256, 0, stream>>>(lin1_W, wLin1, nm);
    transpose_to_bf16<<<(nm * HID * HID + 255) / 256, 256, 0, stream>>>(lin2_W, wLin2, nm);
    transpose_to_bf16<<<(nm * HID * HID + 255) / 256, 256, 0, stream>>>(lin_W,  wLin,  nm);
  }

  embed_kernel<<<(N_NODES * HID + 255) / 256, 256, 0, stream>>>(z, emblin_W, emblin_b, h);

  const int nblkNode = (N_NODES + 127) / 128;
  for (int l = 0; l < L_LAYERS; ++l) {
    zero_kernel<<<(N_NODES * HID + 255) / 256, 256, 0, stream>>>(agg, N_NODES * HID);
    // xf = h @ lin1
    node_gemm128<<<nblkNode, 256, GEMM_LDS, stream>>>(
        h, wLin1 + (size_t)l * HID * HID, nullptr, xf, N_NODES, 0);
    // edge filter MLP + gate + gather + scatter-add
    edge_filter_scatter<<<N_EDGES / 128, 256, GEMM_LDS, stream>>>(
        eattr, eidx, elen,
        wFilt + (size_t)l * FILT_LAYERS * HID * HID,
        filt_b + (size_t)l * FILT_LAYERS * HID, xf, agg);
    // x = gelu(agg @ lin2 + b2)   (reuse xf buffer)
    node_gemm128<<<nblkNode, 256, GEMM_LDS, stream>>>(
        agg, wLin2 + (size_t)l * HID * HID, lin2_b + (size_t)l * HID,
        xf, N_NODES, NG_BIAS | NG_GELU);
    // h += x @ lin_W + b
    node_gemm128<<<nblkNode, 256, GEMM_LDS, stream>>>(
        xf, wLin + (size_t)l * HID * HID, lin_b + (size_t)l * HID,
        h, N_NODES, NG_BIAS | NG_ACCUM);
  }
}
